// NN_22359599743360
// MI455X (gfx1250) — compile-verified, matrix operand order checked by reading, not run on hardware
//
#include <hip/hip_runtime.h>
#include <hip/hip_bf16.h>

#define N_NODES 50000
#define N_EDGES 800000
#define N_GRAPHS 2048
#define NODE_F 11
#define HID 128

typedef __attribute__((ext_vector_type(16))) _Float16 v16h;
typedef __attribute__((ext_vector_type(8)))  float    v8f;

// ---------------------------------------------------------------------------
// Per-node scalar projections a[n]=h[n]·We[0:d], b[n]=h[n]·We[d:2d]; zero agg.
// One wave (32 lanes) per node.
// ---------------------------------------------------------------------------
__global__ void proj_kernel(const float* __restrict__ h, int d,
                            const float* __restrict__ We,
                            float* __restrict__ a, float* __restrict__ b,
                            float* __restrict__ agg, int N) {
    int wave = (int)((blockIdx.x * blockDim.x + threadIdx.x) >> 5);
    int lane = threadIdx.x & 31;
    if (wave >= N) return;
    const float* row = h + (size_t)wave * d;
    float sa = 0.f, sb = 0.f;
    for (int k = lane; k < d; k += 32) {
        float v = row[k];
        sa += v * We[k];
        sb += v * We[d + k];
    }
    for (int off = 16; off > 0; off >>= 1) {
        sa += __shfl_xor(sa, off, 32);
        sb += __shfl_xor(sb, off, 32);
    }
    if (lane == 0) { a[wave] = sa; b[wave] = sb; agg[wave] = 0.f; }
}

// ---------------------------------------------------------------------------
// Per-edge message + scatter-add: agg[dst] += relu(a[dst]+b[src]+ea·we+be)
// ---------------------------------------------------------------------------
__global__ void edge_kernel(const int* __restrict__ src, const int* __restrict__ dst,
                            const float* __restrict__ ea, const float* __restrict__ wE,
                            const float* __restrict__ be,
                            const float* __restrict__ a, const float* __restrict__ b,
                            float* __restrict__ agg, int E) {
    int e = (int)(blockIdx.x * blockDim.x + threadIdx.x);
    if (e >= E) return;
    int s  = src[e];
    int dd = dst[e];
    float4 av = *(const float4*)(ea + (size_t)e * 4);
    float m = a[dd] + b[s] + av.x * wE[0] + av.y * wE[1] + av.z * wE[2] + av.w * wE[3] + be[0];
    m = fmaxf(m, 0.f);
    if (m != 0.f)
        (void)__hip_atomic_fetch_add(&agg[dd], m, __ATOMIC_RELAXED, __HIP_MEMORY_SCOPE_AGENT);
}

// ---------------------------------------------------------------------------
// Layer-1 node update (K=11, too small for WMMA): h = relu([x,agg]@Wh + bh)
// ---------------------------------------------------------------------------
__global__ void node_update_small(const float* __restrict__ x,
                                  const float* __restrict__ Wh,
                                  const float* __restrict__ bh,
                                  const float* __restrict__ agg,
                                  float* __restrict__ hout) {
    int n = blockIdx.x;
    int j = threadIdx.x;           // 128 threads
    float s = bh[j];
    const float* xr = x + (size_t)n * NODE_F;
    #pragma unroll
    for (int k = 0; k < NODE_F; ++k) s += xr[k] * Wh[k * HID + j];
    s += agg[n] * Wh[NODE_F * HID + j];
    hout[(size_t)n * HID + j] = fmaxf(s, 0.f);
}

// ---------------------------------------------------------------------------
// Pre-swizzle W[128x128] f32 -> f16 B-fragments for v_wmma_f32_16x16x32_f16.
// Fragment (kt, nt): lane l holds 16 halfs; n = nt*16 + (l&15);
// half = l>>4; elems 0..7 -> K = kt*32 + half*8 + e; elems 8..15 -> +16.
// ---------------------------------------------------------------------------
__global__ void pack_w_f16(const float* __restrict__ W, _Float16* __restrict__ Wpk) {
    int idx = (int)(blockIdx.x * blockDim.x + threadIdx.x);   // 0..16383
    if (idx >= 4 * 8 * 32 * 16) return;
    int e    = idx & 15;
    int lane = (idx >> 4) & 31;
    int nt   = (idx >> 9) & 7;
    int kt   = (idx >> 12) & 3;
    int half = lane >> 4;
    int k = kt * 32 + half * 8 + (e & 7) + ((e >= 8) ? 16 : 0);
    int n = nt * 16 + (lane & 15);
    Wpk[idx] = (_Float16)W[k * HID + n];
}

// ---------------------------------------------------------------------------
// Node update GEMM: Hout = [H,agg] @ Wh + bh (opt relu). One wave per 16-row
// tile computes a 16x128 output strip with 8 WMMA accumulators.
// ---------------------------------------------------------------------------
__global__ void __launch_bounds__(256)
gemm_node_update(const float* __restrict__ H, const _Float16* __restrict__ Wpk,
                 const float* __restrict__ wAgg, const float* __restrict__ bias,
                 const float* __restrict__ agg, float* __restrict__ Hout,
                 int N, int doRelu) {
    int wave = (int)((blockIdx.x * blockDim.x + threadIdx.x) >> 5);
    int lane = threadIdx.x & 31;
    int m0 = wave * 16;
    if (m0 >= N) return;

    int halfw = lane >> 4;
    int l15   = lane & 15;
    const float* arow = H + (size_t)(m0 + l15) * HID;
    const v16h* wfr = (const v16h*)Wpk;

    v8f zero = {};
    v8f acc[8];
    #pragma unroll
    for (int nt = 0; nt < 8; ++nt) acc[nt] = zero;

    #pragma unroll
    for (int kt = 0; kt < 4; ++kt) {
        int kbase = kt * 32 + halfw * 8;
        float4 f0 = *(const float4*)(arow + kbase);
        float4 f1 = *(const float4*)(arow + kbase + 4);
        float4 f2 = *(const float4*)(arow + kbase + 16);
        float4 f3 = *(const float4*)(arow + kbase + 20);
        v16h afrag;
        afrag[0]  = (_Float16)f0.x; afrag[1]  = (_Float16)f0.y;
        afrag[2]  = (_Float16)f0.z; afrag[3]  = (_Float16)f0.w;
        afrag[4]  = (_Float16)f1.x; afrag[5]  = (_Float16)f1.y;
        afrag[6]  = (_Float16)f1.z; afrag[7]  = (_Float16)f1.w;
        afrag[8]  = (_Float16)f2.x; afrag[9]  = (_Float16)f2.y;
        afrag[10] = (_Float16)f2.z; afrag[11] = (_Float16)f2.w;
        afrag[12] = (_Float16)f3.x; afrag[13] = (_Float16)f3.y;
        afrag[14] = (_Float16)f3.z; afrag[15] = (_Float16)f3.w;

        #pragma unroll
        for (int nt = 0; nt < 8; ++nt) {
            v16h bfrag = wfr[(kt * 8 + nt) * 32 + lane];
            acc[nt] = __builtin_amdgcn_wmma_f32_16x16x32_f16(
                false, afrag, false, bfrag, (short)0, acc[nt], false, false);
        }
    }

    float aggv[8];
    #pragma unroll
    for (int r = 0; r < 8; ++r) aggv[r] = agg[m0 + r + 8 * halfw];

    #pragma unroll
    for (int nt = 0; nt < 8; ++nt) {
        int ncol = nt * 16 + l15;
        float wj = wAgg[ncol];
        float bj = bias[ncol];
        #pragma unroll
        for (int r = 0; r < 8; ++r) {
            float v = acc[nt][r] + aggv[r] * wj + bj;
            if (doRelu) v = fmaxf(v, 0.f);
            Hout[(size_t)(m0 + r + 8 * halfw) * HID + ncol] = v;
        }
    }
}

// ---------------------------------------------------------------------------
// Pooling + readout
// ---------------------------------------------------------------------------
__global__ void zero_kernel(float* __restrict__ p, int n) {
    int i = (int)(blockIdx.x * blockDim.x + threadIdx.x);
    if (i < n) p[i] = 0.f;
}

__global__ void pool_kernel(const float* __restrict__ h, const int* __restrict__ batch,
                            float* __restrict__ g, int total) {
    int idx = (int)(blockIdx.x * blockDim.x + threadIdx.x);
    if (idx >= total) return;
    int n = idx >> 7, j = idx & 127;
    (void)__hip_atomic_fetch_add(&g[(size_t)batch[n] * HID + j], h[idx],
                                 __ATOMIC_RELAXED, __HIP_MEMORY_SCOPE_AGENT);
}

__global__ void readout_kernel(const float* __restrict__ g, const float* __restrict__ Wl,
                               const float* __restrict__ bl, const float* __restrict__ Wl2,
                               const float* __restrict__ bl2, float* __restrict__ out) {
    __shared__ float sh[64];
    int gi = blockIdx.x;
    int j  = threadIdx.x;          // 64 threads
    const float* grow = g + (size_t)gi * HID;
    float s = bl[j];
    for (int k = 0; k < HID; ++k) s += grow[k] * Wl[k * 64 + j];
    s = fmaxf(s, 0.f);
    sh[j] = s * Wl2[j];
    __syncthreads();
    for (int off = 32; off > 0; off >>= 1) {
        if (j < off) sh[j] += sh[j + off];
        __syncthreads();
    }
    if (j == 0) out[gi] = sh[0] + bl2[0];
}

// ---------------------------------------------------------------------------
extern "C" void kernel_launch(void* const* d_in, const int* in_sizes, int n_in,
                              void* d_out, int out_size, void* d_ws, size_t ws_size,
                              hipStream_t stream) {
    (void)in_sizes; (void)n_in; (void)out_size; (void)ws_size;
    const float* x     = (const float*)d_in[0];
    const int*   eidx  = (const int*)d_in[1];
    const float* eattr = (const float*)d_in[2];
    const int*   batch = (const int*)d_in[3];
    const float* We1 = (const float*)d_in[4];  const float* be1 = (const float*)d_in[5];
    const float* Wh1 = (const float*)d_in[6];  const float* bh1 = (const float*)d_in[7];
    const float* We2 = (const float*)d_in[8];  const float* be2 = (const float*)d_in[9];
    const float* Wh2 = (const float*)d_in[10]; const float* bh2 = (const float*)d_in[11];
    const float* We3 = (const float*)d_in[12]; const float* be3 = (const float*)d_in[13];
    const float* Wh3 = (const float*)d_in[14]; const float* bh3 = (const float*)d_in[15];
    const float* Wl  = (const float*)d_in[16]; const float* bl  = (const float*)d_in[17];
    const float* Wl2 = (const float*)d_in[18]; const float* bl2 = (const float*)d_in[19];
    float* out = (float*)d_out;

    const int* src = eidx;            // edge_index[0]
    const int* dst = eidx + N_EDGES;  // edge_index[1]

    char* ws = (char*)d_ws;
    float*     hA   = (float*)(ws + 0);            // 25,600,000 B
    float*     hB   = (float*)(ws + 25600000);     // 25,600,000 B
    float*     pa   = (float*)(ws + 51200000);     // 200,000 B
    float*     pb   = (float*)(ws + 51400192);     // 200,000 B
    float*     agg  = (float*)(ws + 51600384);     // 200,000 B
    float*     g    = (float*)(ws + 51800576);     // 1,048,576 B
    _Float16*  Wpk2 = (_Float16*)(ws + 52849152);  // 32,768 B
    _Float16*  Wpk3 = (_Float16*)(ws + 52881920);  // 32,768 B

    const int projBlocks  = (N_NODES * 32 + 255) / 256;
    const int edgeBlocks  = (N_EDGES + 255) / 256;
    const int gemmBlocks  = ((N_NODES / 16) * 32 + 255) / 256;
    const int poolBlocks  = (N_NODES * HID + 255) / 256;
    const int gzBlocks    = (N_GRAPHS * HID + 255) / 256;

    // ---- Layer 1: in=11 -> 128, relu ----
    proj_kernel<<<projBlocks, 256, 0, stream>>>(x, NODE_F, We1, pa, pb, agg, N_NODES);
    edge_kernel<<<edgeBlocks, 256, 0, stream>>>(src, dst, eattr, We1 + 2 * NODE_F, be1,
                                                pa, pb, agg, N_EDGES);
    node_update_small<<<N_NODES, HID, 0, stream>>>(x, Wh1, bh1, agg, hA);

    // ---- Layer 2: 128 -> 128, relu (WMMA) ----
    pack_w_f16<<<64, 256, 0, stream>>>(Wh2, Wpk2);
    proj_kernel<<<projBlocks, 256, 0, stream>>>(hA, HID, We2, pa, pb, agg, N_NODES);
    edge_kernel<<<edgeBlocks, 256, 0, stream>>>(src, dst, eattr, We2 + 2 * HID, be2,
                                                pa, pb, agg, N_EDGES);
    gemm_node_update<<<gemmBlocks, 256, 0, stream>>>(hA, Wpk2, Wh2 + HID * HID, bh2,
                                                     agg, hB, N_NODES, 1);

    // ---- Layer 3: 128 -> 128, no relu (WMMA) ----
    pack_w_f16<<<64, 256, 0, stream>>>(Wh3, Wpk3);
    proj_kernel<<<projBlocks, 256, 0, stream>>>(hB, HID, We3, pa, pb, agg, N_NODES);
    edge_kernel<<<edgeBlocks, 256, 0, stream>>>(src, dst, eattr, We3 + 2 * HID, be3,
                                                pa, pb, agg, N_EDGES);
    gemm_node_update<<<gemmBlocks, 256, 0, stream>>>(hB, Wpk3, Wh3 + HID * HID, bh3,
                                                     agg, hA, N_NODES, 0);

    // ---- Pool + readout ----
    zero_kernel<<<gzBlocks, 256, 0, stream>>>(g, N_GRAPHS * HID);
    pool_kernel<<<poolBlocks, 256, 0, stream>>>(hA, batch, g, N_NODES * HID);
    readout_kernel<<<N_GRAPHS, 64, 0, stream>>>(g, Wl, bl, Wl2, bl2, out);
}
